// GCN_3616362463929
// MI455X (gfx1250) — matrix operand order repaired and checked by reading, not compiled, not run on hardware
//
#include <hip/hip_runtime.h>
#include <hip/hip_bf16.h>

typedef __attribute__((ext_vector_type(16))) _Float16 v16h;
typedef __attribute__((ext_vector_type(2)))  _Float16 v2h;
typedef __attribute__((ext_vector_type(8)))  float    v8f;

#define WMMA_F16(a, b, c) \
  __builtin_amdgcn_wmma_f32_16x16x32_f16(false, (a), false, (b), (short)0, (c), false, false)

__device__ __forceinline__ float sigf(float x) { return 1.0f / (1.0f + __expf(-x)); }
__device__ __forceinline__ float leakyf(float x) { return x >= 0.0f ? x : 0.01f * x; }
__device__ __forceinline__ float rs_clip(float d) { if (d < 1.0f) d = 1.0f; return rsqrtf(d); }

// ---- WMMA fragment helpers (16x16x32 f16, wave32; contiguous k order used by this file) ----
// A (16x32 f16): lane<16 -> row=lane, K k0..k0+15
//                lane>=16 -> row=lane-16, K k0+16..k0+31
__device__ __forceinline__ v16h load_A_frag(const _Float16* a, int stride, int lane, int k0) {
  int row = lane & 15;
  int kk  = k0 + ((lane >> 4) << 4);
  const _Float16* p = a + row * stride + kk;
  v16h f;
#pragma unroll
  for (int i = 0; i < 8; ++i) f[i] = p[i];
#pragma unroll
  for (int i = 0; i < 8; ++i) f[8 + i] = p[8 + i];
  return f;
}

// B (32x16 f16) from packed BT[n*K + k] (column-major of B): lane<16 -> col=lane, K k0..k0+15
//                                                            lane>=16 -> col=lane-16, K k0+16..k0+31
__device__ __forceinline__ v16h load_B_frag(const _Float16* bt, int K, int lane, int n0, int k0) {
  int n  = n0 + (lane & 15);
  int kk = k0 + ((lane >> 4) << 4);
  const _Float16* p = bt + n * K + kk;
  v16h f;
#pragma unroll
  for (int i = 0; i < 16; ++i) f[i] = p[i];
  return f;
}

// ---------------------------------------------------------------------------
// Utility kernels
// ---------------------------------------------------------------------------
__global__ void zero_kernel(float* p, long long n) {
  long long i = (long long)blockIdx.x * blockDim.x + threadIdx.x;
  if (i < n) p[i] = 0.0f;
}

// Pack fp32 weight into f16 "BT" layout: out[n*K + k] = B[k][n]
// mode 0: in is (K x Nc) row-major  (B = W)          -> in[k*Nc + n]
// mode 1: in is (Nc x K) row-major  (B = W^T usage)  -> in[n*K  + k]
__global__ void pack_b_kernel(const float* in, _Float16* out, int K, int Nc, int mode) {
  int i = blockIdx.x * blockDim.x + threadIdx.x;
  if (i >= K * Nc) return;
  int n = i / K, k = i - n * K;
  float v = mode ? in[n * K + k] : in[k * Nc + n];
  out[i] = (_Float16)v;
}

__global__ void degree_kernel(const int* idx, float* deg, int nE) {
  int e = blockIdx.x * blockDim.x + threadIdx.x;
  if (e < nE) atomicAdd(&deg[idx[e]], 1.0f);
}

// ---------------------------------------------------------------------------
// Bi-LSTM final-state kernel. One block = 16 sequences, 8 waves.
// Embedding table lives in LDS (f16); gate GEMMs (K=64, N=256) done with WMMA,
// weight B-fragments are register-resident across all 32 timesteps.
// ---------------------------------------------------------------------------
__global__ __launch_bounds__(256) void lstm_kernel(
    const int* __restrict__ tokens, const int* __restrict__ maskI,
    const float* __restrict__ emb,  const float* __restrict__ Wih,
    const float* __restrict__ Whh,  const float* __restrict__ bias,
    float* __restrict__ hout, int N, int reverse)
{
  __shared__ _Float16 embL[128 * 64];   // 16KB
  __shared__ _Float16 hL[16 * 64];      // 2KB
  __shared__ float    cL[16 * 64];      // 4KB
  __shared__ float    gL[16 * 256];     // 16KB
  __shared__ int      tokL[16 * 32];    // 2KB
  __shared__ int      mL[16 * 32];      // 2KB

  int tid  = threadIdx.x;
  int lane = tid & 31;
  int wave = tid >> 5;                  // 0..7, each owns 2 N-tiles
  int row0 = blockIdx.x * 16;

  // Packed staging: one float2 load + one 32-bit LDS store per element pair.
  {
    const float2* e2 = (const float2*)emb;
    v2h* d2 = (v2h*)embL;
    for (int i = tid; i < 128 * 64 / 2; i += 256) {
      float2 v = e2[i];
      v2h h2; h2[0] = (_Float16)v.x; h2[1] = (_Float16)v.y;
      d2[i] = h2;
    }
  }
  for (int i = tid; i < 16 * 32; i += 256) {
    int r = i >> 5, t = i & 31;
    int g = row0 + r;
    tokL[i] = (g < N) ? tokens[g * 32 + t] : 0;
    mL[i]   = (g < N) ? maskI[g * 32 + t] : 0;
  }
  for (int i = tid; i < 16 * 64; i += 256) { hL[i] = (_Float16)0.0f; cL[i] = 0.0f; }

  // Register-resident weight B-fragments: 2 n-tiles x 2 k-tiles for Wih and Whh.
  v16h wihB[2][2], whhB[2][2];
  float bV[2];
#pragma unroll
  for (int tn = 0; tn < 2; ++tn) {
    int n0 = (wave * 2 + tn) * 16;
    int n  = n0 + (lane & 15);
    int kk = (lane >> 4) << 4;
#pragma unroll
    for (int tk = 0; tk < 2; ++tk) {
      const float* pw = Wih + n * 64 + tk * 32 + kk;   // B[k][n] = Wih[n][k], contiguous in k
      const float* ph = Whh + n * 64 + tk * 32 + kk;
#pragma unroll
      for (int i = 0; i < 16; ++i) {
        wihB[tn][tk][i] = (_Float16)pw[i];
        whhB[tn][tk][i] = (_Float16)ph[i];
      }
    }
    bV[tn] = bias[n];
  }
  __syncthreads();

  for (int step = 0; step < 32; ++step) {
    int t = reverse ? (31 - step) : step;
#pragma unroll
    for (int tn = 0; tn < 2; ++tn) {
      v8f acc;
#pragma unroll
      for (int i = 0; i < 8; ++i) acc[i] = bV[tn];
#pragma unroll
      for (int tk = 0; tk < 2; ++tk) {
        // A from embedding table (token-indirect), same contiguous k order as load_A_frag and wihB.
        int ar  = lane & 15;
        int tok = tokL[ar * 32 + t];
        int kk  = tk * 32 + ((lane >> 4) << 4);
        const _Float16* pe = embL + tok * 64 + kk;
        v16h ax;
#pragma unroll
        for (int i = 0; i < 8; ++i) ax[i] = pe[i];
#pragma unroll
        for (int i = 0; i < 8; ++i) ax[8 + i] = pe[8 + i];
        acc = WMMA_F16(ax, wihB[tn][tk], acc);

        v16h ah = load_A_frag(hL, 64, lane, tk * 32);
        acc = WMMA_F16(ah, whhB[tn][tk], acc);
      }
      int col = (wave * 2 + tn) * 16 + (lane & 15);
      int r0  = (lane >> 4) << 3;
#pragma unroll
      for (int r = 0; r < 8; ++r) gL[(r0 + r) * 256 + col] = acc[r];
    }
    __syncthreads();
    // Elementwise LSTM cell update with mask.
    for (int i = tid; i < 16 * 64; i += 256) {
      int r = i >> 6, j = i & 63;
      float ig = gL[r * 256 + j];
      float fg = gL[r * 256 + 64 + j];
      float gg = gL[r * 256 + 128 + j];
      float og = gL[r * 256 + 192 + j];
      float cn = sigf(fg) * cL[i] + sigf(ig) * tanhf(gg);
      float hn = sigf(og) * tanhf(cn);
      if (mL[r * 32 + t]) { cL[i] = cn; hL[i] = (_Float16)hn; }
    }
    __syncthreads();
  }
  for (int i = tid; i < 16 * 64; i += 256) {
    int r = i >> 6, j = i & 63;
    int g = row0 + r;
    if (g < N) hout[g * 64 + j] = (float)hL[i];
  }
}

// ---------------------------------------------------------------------------
// FC + concat: h0[:,0:64] = leaky([hf|hb] @ fc_W^T + fc_b), h0[:,64:256] = embeddings.
// ---------------------------------------------------------------------------
__global__ __launch_bounds__(128) void fc_concat_kernel(
    const float* __restrict__ hf, const float* __restrict__ hb,
    const _Float16* __restrict__ fcBT, const float* __restrict__ fcb,
    const int* __restrict__ cat, const int* __restrict__ country, const int* __restrict__ slot,
    const float* __restrict__ embCat, const float* __restrict__ embCo, const float* __restrict__ embSl,
    float* __restrict__ h0, int N)
{
  __shared__ _Float16 aL[16 * 128];     // 4KB
  int tid = threadIdx.x, lane = tid & 31, wave = tid >> 5;   // 4 waves, Nc=64
  int row0 = blockIdx.x * 16;

  // Packed staging of A = [hf|hb] (boundary 64 is even: pairs never straddle).
  {
    v2h* d2 = (v2h*)aL;
    for (int i = tid; i < 16 * 64; i += 128) {   // 16 rows x 64 pairs
      int r = i >> 6, kp = i & 63;
      int k = kp * 2;
      int g = row0 + r;
      float2 v = make_float2(0.0f, 0.0f);
      if (g < N) {
        const float* src = (k < 64) ? (hf + (long long)g * 64 + k)
                                    : (hb + (long long)g * 64 + (k - 64));
        v = *(const float2*)src;
      }
      v2h h2; h2[0] = (_Float16)v.x; h2[1] = (_Float16)v.y;
      d2[i] = h2;
    }
  }
  __syncthreads();

  int n0 = wave * 16;
  int nn = n0 + (lane & 15);
  v8f acc;
#pragma unroll
  for (int i = 0; i < 8; ++i) acc[i] = fcb[nn];
#pragma unroll
  for (int tk = 0; tk < 4; ++tk) {
    v16h a = load_A_frag(aL, 128, lane, tk * 32);
    v16h b = load_B_frag(fcBT, 128, lane, n0, tk * 32);
    acc = WMMA_F16(a, b, acc);
  }
  int r0 = (lane >> 4) << 3;
#pragma unroll
  for (int r = 0; r < 8; ++r) {
    int g = row0 + r0 + r;
    if (g < N) h0[g * 256 + nn] = leakyf(acc[r]);
  }
  // Fill cols 64..255 from categorical embeddings.
  for (int i = tid; i < 16 * 192; i += 128) {
    int r = i / 192, j = i - r * 192;
    int g = row0 + r;
    if (g >= N) continue;
    float v;
    if (j < 64)       v = embCat[cat[g] * 64 + j];
    else if (j < 128) v = embCo[country[g] * 64 + (j - 64)];
    else              v = embSl[slot[g] * 64 + (j - 128)];
    h0[g * 256 + 64 + j] = v;
  }
}

// ---------------------------------------------------------------------------
// Generic node GEMM: out(N x 128) = (A * rsqrt(clip(outdeg,1)))(N x K) @ B(K x 128)
// outdeg == nullptr -> no scaling. 8 waves, one 16x16 tile each.
// ---------------------------------------------------------------------------
template <int K>
__global__ __launch_bounds__(256) void gemm_rows_kernel(
    const float* __restrict__ A, const float* __restrict__ outdeg,
    const _Float16* __restrict__ BT, float* __restrict__ out, int N)
{
  __shared__ _Float16 aL[16 * K];
  int tid = threadIdx.x, lane = tid & 31, wave = tid >> 5;
  int row0 = blockIdx.x * 16;

  {
    v2h* d2 = (v2h*)aL;
    for (int i = tid; i < 16 * K / 2; i += 256) {
      int r = i / (K / 2), kp = i - r * (K / 2);
      int k = kp * 2;
      int g = row0 + r;
      float2 v = make_float2(0.0f, 0.0f);
      if (g < N) {
        v = *(const float2*)(A + (long long)g * K + k);
        float s = outdeg ? rs_clip(outdeg[g]) : 1.0f;
        v.x *= s; v.y *= s;
      }
      v2h h2; h2[0] = (_Float16)v.x; h2[1] = (_Float16)v.y;
      d2[i] = h2;
    }
  }
  __syncthreads();

  int n0 = wave * 16;
  v8f acc;
#pragma unroll
  for (int i = 0; i < 8; ++i) acc[i] = 0.0f;
#pragma unroll
  for (int tk = 0; tk < K / 32; ++tk) {
    v16h a = load_A_frag(aL, K, lane, tk * 32);
    v16h b = load_B_frag(BT, K, lane, n0, tk * 32);
    acc = WMMA_F16(a, b, acc);
  }
  int col = n0 + (lane & 15);
  int r0  = (lane >> 4) << 3;
#pragma unroll
  for (int r = 0; r < 8; ++r) {
    int g = row0 + r0 + r;
    if (g < N) out[(long long)g * 128 + col] = acc[r];
  }
}

// ---------------------------------------------------------------------------
// Edge scatter-add: agg[dst] += feat[src] * (outdeg ? rsqrt(clip(outdeg[src],1)) : 1)
// 32 threads per edge, 4 floats each (128 features).
// ---------------------------------------------------------------------------
__global__ __launch_bounds__(256) void scatter_add_kernel(
    const float* __restrict__ feat, const int* __restrict__ src, const int* __restrict__ dst,
    const float* __restrict__ outdeg, float* __restrict__ agg, int nE)
{
  long long idx = (long long)blockIdx.x * blockDim.x + threadIdx.x;
  int e = (int)(idx >> 5);
  int l = (int)(idx & 31);
  if (e >= nE) return;
  int s = src[e], d = dst[e];
  float sc = outdeg ? rs_clip(outdeg[s]) : 1.0f;
  const float* ps = feat + (long long)s * 128 + l * 4;
  float* pd = agg + (long long)d * 128 + l * 4;
#pragma unroll
  for (int j = 0; j < 4; ++j) atomicAdd(&pd[j], ps[j] * sc);
}

// Combine two relations: out = leaky(aggS*rsqrt(indegS)+bS + aggU*rsqrt(indegU)+bU)
__global__ void combine_kernel(
    const float* __restrict__ aggS, const float* __restrict__ aggU,
    const float* __restrict__ indegS, const float* __restrict__ indegU,
    const float* __restrict__ bS, const float* __restrict__ bU,
    float* __restrict__ out, int N)
{
  long long i = (long long)blockIdx.x * blockDim.x + threadIdx.x;
  if (i >= (long long)N * 128) return;
  int g = (int)(i >> 7), j = (int)(i & 127);
  float v = aggS[i] * rs_clip(indegS[g]) + bS[j] + aggU[i] * rs_clip(indegU[g]) + bU[j];
  out[i] = leakyf(v);
}

// ---------------------------------------------------------------------------
// Classifier GEMM: z(nE x 128) = [h[src]|h[dst]](nE x 256) @ cls_W1^T + cls_b1
// ---------------------------------------------------------------------------
__global__ __launch_bounds__(256) void edge_gemm_kernel(
    const float* __restrict__ h, const int* __restrict__ srcI, const int* __restrict__ dstI,
    const _Float16* __restrict__ w1BT, const float* __restrict__ b1,
    float* __restrict__ z, int nE)
{
  __shared__ _Float16 aL[16 * 256];     // 8KB
  int tid = threadIdx.x, lane = tid & 31, wave = tid >> 5;
  int e0 = blockIdx.x * 16;

  // Packed gather staging (boundary 128 is even: pairs never straddle).
  {
    v2h* d2 = (v2h*)aL;
    for (int i = tid; i < 16 * 128; i += 256) {   // 16 rows x 128 pairs
      int r = i >> 7, kp = i & 127;
      int k = kp * 2;
      int e = e0 + r;
      float2 v = make_float2(0.0f, 0.0f);
      if (e < nE) {
        const float* src = (k < 128) ? (h + (long long)srcI[e] * 128 + k)
                                     : (h + (long long)dstI[e] * 128 + (k - 128));
        v = *(const float2*)src;
      }
      v2h h2; h2[0] = (_Float16)v.x; h2[1] = (_Float16)v.y;
      d2[i] = h2;
    }
  }
  __syncthreads();

  int n0 = wave * 16;
  int nn = n0 + (lane & 15);
  v8f acc;
#pragma unroll
  for (int i = 0; i < 8; ++i) acc[i] = b1[nn];
#pragma unroll
  for (int tk = 0; tk < 8; ++tk) {
    v16h a = load_A_frag(aL, 256, lane, tk * 32);
    v16h b = load_B_frag(w1BT, 256, lane, n0, tk * 32);
    acc = WMMA_F16(a, b, acc);
  }
  int r0 = (lane >> 4) << 3;
#pragma unroll
  for (int r = 0; r < 8; ++r) {
    int e = e0 + r0 + r;
    if (e < nE) z[(long long)e * 128 + nn] = acc[r];
  }
}

// BatchNorm stats over batch dim: one block per feature column.
// Strided (512B) column walk over a row-major array -> prefetch one
// grid-stride ahead (lowers to global_prefetch_b8 on gfx1250).
__global__ __launch_bounds__(256) void bn_stats_kernel(
    const float* __restrict__ z, float* __restrict__ meanv, float* __restrict__ varv, int nE)
{
  int j = blockIdx.x;
  float s = 0.0f, s2 = 0.0f;
  for (int e = threadIdx.x; e < nE; e += 256) {
    const float* p = &z[(long long)e * 128 + j];
    if (e + 256 < nE) __builtin_prefetch(p + (long long)256 * 128, 0, 1);
    float v = *p;
    s += v; s2 += v * v;
  }
  __shared__ float rs[256], rq[256];
  rs[threadIdx.x] = s; rq[threadIdx.x] = s2;
  __syncthreads();
  for (int o = 128; o > 0; o >>= 1) {
    if (threadIdx.x < o) { rs[threadIdx.x] += rs[threadIdx.x + o]; rq[threadIdx.x] += rq[threadIdx.x + o]; }
    __syncthreads();
  }
  if (threadIdx.x == 0) {
    float m = rs[0] / nE;
    meanv[j] = m;
    varv[j]  = rq[0] / nE - m * m;
  }
}

// Normalize + relu + 128->2 head.
__global__ void final_kernel(
    const float* __restrict__ z, const float* __restrict__ meanv, const float* __restrict__ varv,
    const float* __restrict__ bng, const float* __restrict__ bnb,
    const float* __restrict__ W2, const float* __restrict__ b2,
    float* __restrict__ out, int nE)
{
  int e = blockIdx.x * blockDim.x + threadIdx.x;
  if (e >= nE) return;
  float a0 = b2[0], a1 = b2[1];
  const float* zr = z + (long long)e * 128;
#pragma unroll 4
  for (int j = 0; j < 128; ++j) {
    float v = (zr[j] - meanv[j]) * rsqrtf(varv[j] + 1e-5f) * bng[j] + bnb[j];
    v = fmaxf(v, 0.0f);
    a0 += v * W2[j];
    a1 += v * W2[128 + j];
  }
  out[e * 2 + 0] = a0;
  out[e * 2 + 1] = a1;
}

// ---------------------------------------------------------------------------
extern "C" void kernel_launch(void* const* d_in, const int* in_sizes, int n_in,
                              void* d_out, int out_size, void* d_ws, size_t ws_size,
                              hipStream_t stream) {
  const int* inputs_s   = (const int*)d_in[0];
  const int* inputs_sm  = (const int*)d_in[1];
  const int* inputs_c   = (const int*)d_in[2];
  const int* inputs_co  = (const int*)d_in[3];
  const int* inputs_sl  = (const int*)d_in[4];
  const int* sim_src    = (const int*)d_in[5];
  const int* sim_dst    = (const int*)d_in[6];
  const int* user_src   = (const int*)d_in[7];
  const int* user_dst   = (const int*)d_in[8];
  const int* esub_src   = (const int*)d_in[9];
  const int* esub_dst   = (const int*)d_in[10];
  const float* emb_url  = (const float*)d_in[11];
  const float* emb_cat  = (const float*)d_in[12];
  const float* emb_co   = (const float*)d_in[13];
  const float* emb_sl   = (const float*)d_in[14];
  const float* Wih_f    = (const float*)d_in[15];
  const float* Whh_f    = (const float*)d_in[16];
  const float* b_f      = (const float*)d_in[17];
  const float* Wih_b    = (const float*)d_in[18];
  const float* Whh_b    = (const float*)d_in[19];
  const float* b_b      = (const float*)d_in[20];
  const float* fc_W     = (const float*)d_in[21];
  const float* fc_b     = (const float*)d_in[22];
  const float* g0s_W    = (const float*)d_in[23];
  const float* g0s_b    = (const float*)d_in[24];
  const float* g0u_W    = (const float*)d_in[25];
  const float* g0u_b    = (const float*)d_in[26];
  const float* g1s_W    = (const float*)d_in[27];
  const float* g1s_b    = (const float*)d_in[28];
  const float* g1u_W    = (const float*)d_in[29];
  const float* g1u_b    = (const float*)d_in[30];
  const float* cls_W1   = (const float*)d_in[31];
  const float* cls_b1   = (const float*)d_in[32];
  const float* bn_g     = (const float*)d_in[33];
  const float* bn_b     = (const float*)d_in[34];
  const float* cls_W2   = (const float*)d_in[35];
  const float* cls_b2   = (const float*)d_in[36];

  const int N    = in_sizes[2];
  const int nSim = in_sizes[5];
  const int nUsr = in_sizes[7];
  const int nSub = in_sizes[9];

  // ---- workspace carve-out (bytes, 256-aligned) ----
  char* base = (char*)d_ws;
  size_t cur = 0;
  auto alloc = [&](size_t bytes) {
    char* p = base + cur;
    cur += (bytes + 255) & ~(size_t)255;
    return p;
  };
  float* HF = (float*)alloc((size_t)N * 64 * 4);
  float* HB = (float*)alloc((size_t)N * 64 * 4);
  float* H0 = (float*)alloc((size_t)N * 256 * 4);   // also reused as a 128-wide scratch later
  float* S0 = (float*)alloc((size_t)N * 128 * 4);
  float* S1 = (float*)alloc((size_t)N * 128 * 4);
  float* S2 = (float*)alloc((size_t)N * 128 * 4);
  float* S3 = (float*)alloc((size_t)N * 128 * 4);
  float* Z  = (float*)alloc((size_t)nSub * 128 * 4);
  float* DEG = (float*)alloc((size_t)4 * N * 4);
  float* dS_out = DEG, *dS_in = DEG + N, *dU_out = DEG + 2 * N, *dU_in = DEG + 3 * N;
  float* BNm = (float*)alloc(256 * 4);
  float* BNv = BNm + 128;
  _Float16* fcBT  = (_Float16*)alloc(64 * 128 * 2);
  _Float16* g0sBT = (_Float16*)alloc(256 * 128 * 2);
  _Float16* g0uBT = (_Float16*)alloc(256 * 128 * 2);
  _Float16* g1sBT = (_Float16*)alloc(128 * 128 * 2);
  _Float16* g1uBT = (_Float16*)alloc(128 * 128 * 2);
  _Float16* w1BT  = (_Float16*)alloc(128 * 256 * 2);
  (void)ws_size; (void)n_in; (void)out_size;

  const int nBlk16 = (N + 15) / 16;

  // ---- pack weights to f16 BT layout ----
  pack_b_kernel<<<(64 * 128 + 255) / 256, 256, 0, stream>>>(fc_W, fcBT, 128, 64, 1);
  pack_b_kernel<<<(256 * 128 + 255) / 256, 256, 0, stream>>>(g0s_W, g0sBT, 256, 128, 0);
  pack_b_kernel<<<(256 * 128 + 255) / 256, 256, 0, stream>>>(g0u_W, g0uBT, 256, 128, 0);
  pack_b_kernel<<<(128 * 128 + 255) / 256, 256, 0, stream>>>(g1s_W, g1sBT, 128, 128, 0);
  pack_b_kernel<<<(128 * 128 + 255) / 256, 256, 0, stream>>>(g1u_W, g1uBT, 128, 128, 0);
  pack_b_kernel<<<(128 * 256 + 255) / 256, 256, 0, stream>>>(cls_W1, w1BT, 256, 128, 1);

  // ---- Bi-LSTM ----
  lstm_kernel<<<nBlk16, 256, 0, stream>>>(inputs_s, inputs_sm, emb_url, Wih_f, Whh_f, b_f, HF, N, 0);
  lstm_kernel<<<nBlk16, 256, 0, stream>>>(inputs_s, inputs_sm, emb_url, Wih_b, Whh_b, b_b, HB, N, 1);

  // ---- FC + concat -> H0 (N x 256) ----
  fc_concat_kernel<<<nBlk16, 128, 0, stream>>>(HF, HB, fcBT, fc_b, inputs_c, inputs_co, inputs_sl,
                                               emb_cat, emb_co, emb_sl, H0, N);

  // ---- degrees ----
  zero_kernel<<<(4 * N + 255) / 256, 256, 0, stream>>>(DEG, (long long)4 * N);
  degree_kernel<<<(nSim + 255) / 256, 256, 0, stream>>>(sim_src, dS_out, nSim);
  degree_kernel<<<(nSim + 255) / 256, 256, 0, stream>>>(sim_dst, dS_in, nSim);
  degree_kernel<<<(nUsr + 255) / 256, 256, 0, stream>>>(user_src, dU_out, nUsr);
  degree_kernel<<<(nUsr + 255) / 256, 256, 0, stream>>>(user_dst, dU_in, nUsr);

  const long long NF = (long long)N * 128;
  const int zb = (int)((NF + 255) / 256);
  const int scatS = (int)(((long long)nSim * 32 + 255) / 256);
  const int scatU = (int)(((long long)nUsr * 32 + 255) / 256);

  // ---- GCN layer 0 (256 -> 128): transform then scatter ----
  gemm_rows_kernel<256><<<nBlk16, 256, 0, stream>>>(H0, dS_out, g0sBT, S0, N);
  zero_kernel<<<zb, 256, 0, stream>>>(S1, NF);
  scatter_add_kernel<<<scatS, 256, 0, stream>>>(S0, sim_src, sim_dst, nullptr, S1, nSim);
  gemm_rows_kernel<256><<<nBlk16, 256, 0, stream>>>(H0, dU_out, g0uBT, S0, N);
  zero_kernel<<<zb, 256, 0, stream>>>(S2, NF);
  scatter_add_kernel<<<scatU, 256, 0, stream>>>(S0, user_src, user_dst, nullptr, S2, nUsr);
  combine_kernel<<<zb, 256, 0, stream>>>(S1, S2, dS_in, dU_in, g0s_b, g0u_b, S3, N);  // h1 -> S3

  // ---- GCN layer 1 (128 -> 128): scatter (with src scaling) then transform ----
  zero_kernel<<<zb, 256, 0, stream>>>(S0, NF);
  scatter_add_kernel<<<scatS, 256, 0, stream>>>(S3, sim_src, sim_dst, dS_out, S0, nSim);
  zero_kernel<<<zb, 256, 0, stream>>>(S1, NF);
  scatter_add_kernel<<<scatU, 256, 0, stream>>>(S3, user_src, user_dst, dU_out, S1, nUsr);
  gemm_rows_kernel<128><<<nBlk16, 256, 0, stream>>>(S0, nullptr, g1sBT, S2, N);
  float* T = H0;  // H0 dead after layer-0 GEMMs; reuse first N*128 floats
  gemm_rows_kernel<128><<<nBlk16, 256, 0, stream>>>(S1, nullptr, g1uBT, T, N);
  combine_kernel<<<zb, 256, 0, stream>>>(S2, T, dS_in, dU_in, g1s_b, g1u_b, S3, N);   // h2 -> S3

  // ---- edge classifier ----
  edge_gemm_kernel<<<(nSub + 15) / 16, 256, 0, stream>>>(S3, esub_src, esub_dst, w1BT, cls_b1, Z, nSub);
  bn_stats_kernel<<<128, 256, 0, stream>>>(Z, BNm, BNv, nSub);
  final_kernel<<<(nSub + 255) / 256, 256, 0, stream>>>(Z, BNm, BNv, bn_g, bn_b, cls_W2, cls_b2,
                                                       (float*)d_out, nSub);
}
